// VectorQuantizerEMA_49615462203699
// MI455X (gfx1250) — matrix operand order repaired and checked by reading, not compiled, not run on hardware
//
#include <hip/hip_runtime.h>
#include <hip/hip_bf16.h>

#define NUM_EMB 8192
#define EMB_DIM 512
#define BATCH   32768
#define DECAY_F 0.99f
#define OMD_F   0.01f
#define EPS_F   1e-5f

#define ROWS_PER_BLK 128     // 8 waves x 16 rows
#define NT 64                // embedding columns per LDS tile
#define LDA 520              // padded halves per LDS row (1040 B)
#define TILE_HALVES (NT * LDA)           // 33,280 halves = 66,560 B per buffer

typedef __attribute__((ext_vector_type(16))) __bf16 v16bf;
typedef __attribute__((ext_vector_type(8)))  float  v8f;

struct U32x8 { uint4 lo, hi; };

static __device__ __forceinline__ unsigned short f32_to_bf16(float f) {
  unsigned int u = __float_as_uint(f);
  u += 0x7fffu + ((u >> 16) & 1u);     // round-to-nearest-even
  return (unsigned short)(u >> 16);
}
static __device__ __forceinline__ unsigned int pack_bf16(float a, float b) {
  return (unsigned int)f32_to_bf16(a) | ((unsigned int)f32_to_bf16(b) << 16);
}

// ---------------------------------------------------------------------------
// Kernel 1: E fp32 -> bf16 (workspace) + ||E_k||^2
// ---------------------------------------------------------------------------
__global__ void __launch_bounds__(256)
prep_embed_kernel(const float* __restrict__ E,
                  unsigned int* __restrict__ Ebf_u32,
                  float* __restrict__ enorm) {
  const int k = blockIdx.x;
  const int t = threadIdx.x;
  const float2 e2 = ((const float2*)(E + (size_t)k * EMB_DIM))[t];
  Ebf_u32[(size_t)k * (EMB_DIM / 2) + t] = pack_bf16(e2.x, e2.y);
  float s = e2.x * e2.x + e2.y * e2.y;
  for (int o = 16; o > 0; o >>= 1) s += __shfl_down(s, o, 32);
  __shared__ float red[8];
  const int lane = t & 31, w = t >> 5;
  if (lane == 0) red[w] = s;
  __syncthreads();
  if (t == 0) {
    float tot = 0.f;
    #pragma unroll
    for (int i = 0; i < 8; ++i) tot += red[i];
    enorm[k] = tot;
  }
}

// ---------------------------------------------------------------------------
// Kernel 2: fused bf16-WMMA distance GEMM + per-row argmin
// 8 waves x 16 rows; E tiles double-buffered in LDS, staged with
// GLOBAL_LOAD_ASYNC_TO_LDS_B128 (ASYNCcnt) overlapped with WMMA compute.
// B fragments software-pipelined one k-step ahead to deepen DS pipelining.
// ---------------------------------------------------------------------------
__global__ void __launch_bounds__(256)
dist_argmin_kernel(const float* __restrict__ z,
                   const unsigned short* __restrict__ Ebf,
                   const float* __restrict__ enorm,
                   int* __restrict__ indices) {
  extern __shared__ unsigned short et[];     // 2 x [NT][LDA]

  const int t = threadIdx.x;
  const size_t rowBase = (size_t)blockIdx.x * ROWS_PER_BLK;
  const int lane    = t & 31;
  const int wave    = t >> 5;
  const int lane_m  = lane & 15;             // A-row / B-col / C-col within tile
  const int lane_hi = lane >> 4;             // K sub-block select
  const int m0      = wave * 16;

  // ---- Load all 16 A fragments (full D=512) for this wave's rows, keep in VGPRs
  v16bf afrag[16];
  {
    const float* zrow = z + (rowBase + m0 + lane_m) * (size_t)EMB_DIM;
    #pragma unroll
    for (int kk = 0; kk < 16; ++kk) {
      const int kb = kk * 32 + lane_hi * 8;
      float4 f0 = *(const float4*)(zrow + kb);
      float4 f1 = *(const float4*)(zrow + kb + 4);
      float4 f2 = *(const float4*)(zrow + kb + 16);
      float4 f3 = *(const float4*)(zrow + kb + 20);
      U32x8 u;
      u.lo.x = pack_bf16(f0.x, f0.y); u.lo.y = pack_bf16(f0.z, f0.w);
      u.lo.z = pack_bf16(f1.x, f1.y); u.lo.w = pack_bf16(f1.z, f1.w);
      u.hi.x = pack_bf16(f2.x, f2.y); u.hi.y = pack_bf16(f2.z, f2.w);
      u.hi.z = pack_bf16(f3.x, f3.y); u.hi.w = pack_bf16(f3.z, f3.w);
      afrag[kk] = __builtin_bit_cast(v16bf, u);
    }
  }

  float minv[8];
  int   mini[8];
  #pragma unroll
  for (int r = 0; r < 8; ++r) { minv[r] = 3.402823466e+38f; mini[r] = 0; }

  const unsigned etBase = (unsigned)(uintptr_t)(void*)et;  // LDS byte offset (addr[31:0])

  // async-stage one 64-row E tile into LDS buffer `buf` (no VGPR data round-trip)
  auto stage_async = [&](int n0, int buf) {
    const char* src = (const char*)(Ebf + (size_t)n0 * EMB_DIM);
    const unsigned bufBase = etBase + (unsigned)buf * (TILE_HALVES * 2u);
    #pragma unroll
    for (int i = 0; i < 16; ++i) {
      const int c   = t + 256 * i;            // 16-byte chunk id, 64 chunks/row
      const int row = c >> 6;
      const int col = (c & 63) << 3;          // halves
      const unsigned lds_addr = bufBase + (unsigned)(row * LDA + col) * 2u;
      const void* g = src + (size_t)c * 16;
      asm volatile("global_load_async_to_lds_b128 %0, %1, off"
                   :: "v"(lds_addr), "v"(g) : "memory");
    }
  };

  // prologue: stage tile 0
  stage_async(0, 0);
  asm volatile("s_wait_asynccnt 0x0" ::: "memory");
  __syncthreads();

  for (int it = 0; it < NUM_EMB / NT; ++it) {
    const int n0 = it * NT;
    const unsigned short* cur = et + (it & 1) * TILE_HALVES;

    // overlap: stage next tile into the other buffer while we compute
    if (it + 1 < NUM_EMB / NT) stage_async(n0 + NT, (it + 1) & 1);

    v8f acc[4];
    #pragma unroll
    for (int a = 0; a < 4; ++a)
      #pragma unroll
      for (int i = 0; i < 8; ++i) acc[a][i] = 0.f;

    // software-pipelined B fragments: preload kk=0
    U32x8 bcur[4];
    #pragma unroll
    for (int a = 0; a < 4; ++a) {
      const unsigned short* pb = cur + (a * 16 + lane_m) * LDA + lane_hi * 8;
      bcur[a].lo = *(const uint4*)pb;
      bcur[a].hi = *(const uint4*)(pb + 16);
    }

    #pragma unroll
    for (int kk = 0; kk < 16; ++kk) {
      U32x8 bnext[4];
      if (kk < 15) {
        #pragma unroll
        for (int a = 0; a < 4; ++a) {
          const unsigned short* pb =
              cur + (a * 16 + lane_m) * LDA + (kk + 1) * 32 + lane_hi * 8;
          bnext[a].lo = *(const uint4*)pb;
          bnext[a].hi = *(const uint4*)(pb + 16);
        }
      }
      #pragma unroll
      for (int a = 0; a < 4; ++a)
        acc[a] = __builtin_amdgcn_wmma_f32_16x16x32_bf16(
                     false, afrag[kk], false, __builtin_bit_cast(v16bf, bcur[a]),
                     (short)0, acc[a], false, false);
      if (kk < 15) {
        #pragma unroll
        for (int a = 0; a < 4; ++a) bcur[a] = bnext[a];
      }
    }

    #pragma unroll
    for (int a = 0; a < 4; ++a) {
      const int n = n0 + a * 16 + lane_m;
      const float en = enorm[n];
      #pragma unroll
      for (int r = 0; r < 8; ++r) {
        const float d = en - 2.0f * acc[a][r];   // ||E||^2 - 2 z.E  (||z||^2 const/row)
        if (d < minv[r]) { minv[r] = d; mini[r] = n; }
      }
    }

    // staging of next tile must be complete (and this tile's readers done)
    asm volatile("s_wait_asynccnt 0x0" ::: "memory");
    __syncthreads();
  }

  // ---- cross-lane argmin within each 16-lane half (wave32 shuffles)
  #pragma unroll
  for (int r = 0; r < 8; ++r) {
    #pragma unroll
    for (int off = 1; off < 16; off <<= 1) {
      const float ov = __shfl_xor(minv[r], off, 32);
      const int   oi = __shfl_xor(mini[r], off, 32);
      if (ov < minv[r] || (ov == minv[r] && oi < mini[r])) { minv[r] = ov; mini[r] = oi; }
    }
  }
  if (lane_m == 0) {
    #pragma unroll
    for (int r = 0; r < 8; ++r)
      indices[rowBase + m0 + lane_hi * 8 + r] = mini[r];   // C row M = r + 8*lane_hi
  }
}

// ---------------------------------------------------------------------------
// Kernel 3: gather quantized rows (fp32 codebook), MSE loss, counts, dw
// ---------------------------------------------------------------------------
__global__ void __launch_bounds__(128)
gather_kernel(const float* __restrict__ z,
              const float* __restrict__ E,
              const int* __restrict__ indices,
              float* __restrict__ out_q,
              float* __restrict__ counts,
              float* __restrict__ dw,
              float* __restrict__ loss_accum) {
  const int b = blockIdx.x;
  const int t = threadIdx.x;
  const int idx = indices[b];

  const float4 zq = ((const float4*)(z + (size_t)b * EMB_DIM))[t];
  const float4 eq = ((const float4*)(E + (size_t)idx * EMB_DIM))[t];
  ((float4*)(out_q + (size_t)b * EMB_DIM))[t] = eq;   // straight-through == quantized

  const float dx = zq.x - eq.x, dy = zq.y - eq.y, dz = zq.z - eq.z, dw4 = zq.w - eq.w;
  float s = dx * dx + dy * dy + dz * dz + dw4 * dw4;

  float* dwp = dw + (size_t)idx * EMB_DIM + t * 4;
  __hip_atomic_fetch_add(dwp + 0, zq.x, __ATOMIC_RELAXED, __HIP_MEMORY_SCOPE_AGENT);
  __hip_atomic_fetch_add(dwp + 1, zq.y, __ATOMIC_RELAXED, __HIP_MEMORY_SCOPE_AGENT);
  __hip_atomic_fetch_add(dwp + 2, zq.z, __ATOMIC_RELAXED, __HIP_MEMORY_SCOPE_AGENT);
  __hip_atomic_fetch_add(dwp + 3, zq.w, __ATOMIC_RELAXED, __HIP_MEMORY_SCOPE_AGENT);

  for (int o = 16; o > 0; o >>= 1) s += __shfl_down(s, o, 32);
  __shared__ float red[4];
  if ((t & 31) == 0) red[t >> 5] = s;
  __syncthreads();
  if (t == 0) {
    const float tot = red[0] + red[1] + red[2] + red[3];
    __hip_atomic_fetch_add(loss_accum, tot, __ATOMIC_RELAXED, __HIP_MEMORY_SCOPE_AGENT);
    __hip_atomic_fetch_add(&counts[idx], 1.0f, __ATOMIC_RELAXED, __HIP_MEMORY_SCOPE_AGENT);
  }
}

// ---------------------------------------------------------------------------
// Kernel 4: raw new_cluster_size + global sum n
// ---------------------------------------------------------------------------
__global__ void __launch_bounds__(256)
ema_cluster_kernel(const float* __restrict__ ecs,
                   const float* __restrict__ counts,
                   float* __restrict__ ncs_raw,
                   float* __restrict__ n_accum) {
  const int k = blockIdx.x * 256 + threadIdx.x;
  const float raw = DECAY_F * ecs[k] + OMD_F * counts[k];
  ncs_raw[k] = raw;
  float s = raw;
  for (int o = 16; o > 0; o >>= 1) s += __shfl_down(s, o, 32);
  __shared__ float red[8];
  const int lane = threadIdx.x & 31, w = threadIdx.x >> 5;
  if (lane == 0) red[w] = s;
  __syncthreads();
  if (threadIdx.x == 0) {
    float tot = 0.f;
    #pragma unroll
    for (int i = 0; i < 8; ++i) tot += red[i];
    __hip_atomic_fetch_add(n_accum, tot, __ATOMIC_RELAXED, __HIP_MEMORY_SCOPE_AGENT);
  }
}

// ---------------------------------------------------------------------------
// Kernel 5: finalize vq_loss scalar
// ---------------------------------------------------------------------------
__global__ void loss_final_kernel(const float* __restrict__ loss_accum,
                                  float* __restrict__ out_loss) {
  *out_loss = *loss_accum * (1.25f / (float)(BATCH * EMB_DIM));  // (1 + 0.25) * MSE
}

// ---------------------------------------------------------------------------
// Kernel 6: new_ema_w, Laplace-smoothed new_cluster_size, new_embedding_w
// ---------------------------------------------------------------------------
__global__ void __launch_bounds__(256)
ema_final_kernel(const float* __restrict__ ema_w,
                 const float* __restrict__ dw,
                 const float* __restrict__ ncs_raw,
                 const float* __restrict__ n_accum,
                 float* __restrict__ out_emb,
                 float* __restrict__ out_cs,
                 float* __restrict__ out_ema) {
  const int i4 = blockIdx.x * 256 + threadIdx.x;   // float4 index
  const int k  = i4 >> 7;                          // 128 float4 per row
  const float n   = *n_accum;
  const float raw = ncs_raw[k];
  const float ncs = (raw + EPS_F) / (n + EPS_F) * n;
  const float4 w = ((const float4*)ema_w)[i4];
  const float4 d = ((const float4*)dw)[i4];
  float4 ne;
  ne.x = DECAY_F * w.x + OMD_F * d.x;
  ne.y = DECAY_F * w.y + OMD_F * d.y;
  ne.z = DECAY_F * w.z + OMD_F * d.z;
  ne.w = DECAY_F * w.w + OMD_F * d.w;
  ((float4*)out_ema)[i4] = ne;
  const float inv = 1.0f / ncs;
  float4 nb; nb.x = ne.x * inv; nb.y = ne.y * inv; nb.z = ne.z * inv; nb.w = ne.w * inv;
  ((float4*)out_emb)[i4] = nb;
  if ((i4 & 127) == 0) out_cs[k] = ncs;
}

// ---------------------------------------------------------------------------
// Workspace layout (bytes)
// ---------------------------------------------------------------------------
#define WS_EBF      0u           // 8192*512*2      = 8,388,608
#define WS_ENORM    8388608u     // 8192*4          =    32,768
#define WS_IDX      8421376u     // 32768*4         =   131,072
#define WS_COUNTS   8552448u     // 8192*4          =    32,768
#define WS_DW       8585216u     // 8192*512*4     = 16,777,216
#define WS_NCSRAW   25362432u    // 8192*4          =    32,768
#define WS_ACCUM    25395200u    // n_accum, loss_accum (16 B)

extern "C" void kernel_launch(void* const* d_in, const int* in_sizes, int n_in,
                              void* d_out, int out_size, void* d_ws, size_t ws_size,
                              hipStream_t stream) {
  (void)in_sizes; (void)n_in; (void)out_size; (void)ws_size;

  const float* z_e   = (const float*)d_in[0];
  const float* emb_w = (const float*)d_in[1];
  const float* ecs   = (const float*)d_in[2];
  const float* ema_w = (const float*)d_in[3];

  char* ws = (char*)d_ws;
  unsigned short* ebf   = (unsigned short*)(ws + WS_EBF);
  float* enorm          = (float*)(ws + WS_ENORM);
  int*   indices        = (int*)(ws + WS_IDX);
  float* counts         = (float*)(ws + WS_COUNTS);
  float* dw             = (float*)(ws + WS_DW);
  float* ncs_raw        = (float*)(ws + WS_NCSRAW);
  float* n_accum        = (float*)(ws + WS_ACCUM);
  float* loss_accum     = (float*)(ws + WS_ACCUM + 4);

  float* out_q    = (float*)d_out;
  float* out_loss = out_q + (size_t)BATCH * EMB_DIM;
  float* out_emb  = out_loss + 1;
  float* out_cs   = out_emb + (size_t)NUM_EMB * EMB_DIM;
  float* out_ema  = out_cs + NUM_EMB;

  // zero counts + dw (contiguous) and the two accumulators
  hipMemsetAsync(ws + WS_COUNTS, 0, (WS_NCSRAW - WS_COUNTS), stream);
  hipMemsetAsync(ws + WS_ACCUM, 0, 16, stream);

  prep_embed_kernel<<<NUM_EMB, 256, 0, stream>>>(emb_w, (unsigned int*)ebf, enorm);

  const size_t shmem = 2u * TILE_HALVES * sizeof(unsigned short);  // 133,120 B
  dist_argmin_kernel<<<BATCH / ROWS_PER_BLK, 256, shmem, stream>>>(z_e, ebf, enorm, indices);

  gather_kernel<<<BATCH, 128, 0, stream>>>(z_e, emb_w, indices, out_q, counts, dw, loss_accum);

  ema_cluster_kernel<<<NUM_EMB / 256, 256, 0, stream>>>(ecs, counts, ncs_raw, n_accum);

  loss_final_kernel<<<1, 1, 0, stream>>>(loss_accum, out_loss);

  ema_final_kernel<<<(NUM_EMB * EMB_DIM / 4) / 256, 256, 0, stream>>>(
      ema_w, dw, ncs_raw, n_accum, out_emb, out_cs, out_ema);
}
